// FlaxFalconAttention_21251498181511
// MI455X (gfx1250) — matrix operand order repaired and checked by reading, not compiled, not run on hardware
//
#include <hip/hip_runtime.h>
#include <hip/hip_bf16.h>

typedef __attribute__((ext_vector_type(16))) __bf16 bf16x16;
typedef __attribute__((ext_vector_type(8)))  float  f32x8;

union Frag16 {
  bf16x16 bf;
  uint4 q[2];
  unsigned short us[16];
};
union AccF {
  f32x8 v;
  float e[8];
};

__device__ __forceinline__ unsigned short f2bf_u(float x) {
  unsigned u = __float_as_uint(x);
  u += 0x7FFFu + ((u >> 16) & 1u);   // round-to-nearest-even
  return (unsigned short)(u >> 16);
}

// Generic pointers to LDS carry the LDS byte offset in addr[31:0] (ISA 10.2).
__device__ __forceinline__ unsigned lds_u32(const void* p) {
  return (unsigned)(size_t)p;
}

// CDNA5 async copy: 16B global -> LDS, tracked by ASYNCcnt (no VGPR staging).
__device__ __forceinline__ void async_cp16(unsigned lds_byte, const void* gp) {
  asm volatile("global_load_async_to_lds_b128 %0, %1, off"
               :: "v"(lds_byte), "v"((unsigned long long)(size_t)gp)
               : "memory");
}

#define WMMA_BF16(A, B, C) \
  __builtin_amdgcn_wmma_f32_16x16x32_bf16(false, (A), false, (B), (short)0, (C), false, false)

// ---------------------------------------------------------------- conversions

__global__ void cvt_f32_bf16(const float* __restrict__ in,
                             unsigned short* __restrict__ out, long n) {
  long i = (long)blockIdx.x * blockDim.x + threadIdx.x;
  long stride = (long)gridDim.x * blockDim.x;
  for (; i < n; i += stride) out[i] = f2bf_u(in[i]);
}

// in: [K][N] f32, out: [N][K] bf16 (LDS-tiled transpose + convert)
__global__ __launch_bounds__(256) void transp_f32_bf16(
    const float* __restrict__ in, unsigned short* __restrict__ out,
    int K, int N) {
  __shared__ float tile[32][33];
  int n  = blockIdx.x * 32 + threadIdx.x;
  int k0 = blockIdx.y * 32;
#pragma unroll
  for (int j = 0; j < 32; j += 8)
    tile[threadIdx.y + j][threadIdx.x] =
        in[(size_t)(k0 + threadIdx.y + j) * N + n];
  __syncthreads();
  int ko = k0 + threadIdx.x;
#pragma unroll
  for (int j = 0; j < 32; j += 8) {
    int no = blockIdx.x * 32 + threadIdx.y + j;
    out[(size_t)no * K + ko] = f2bf_u(tile[threadIdx.x][threadIdx.y + j]);
  }
}

// ---------------------------------------------------------------- GEMM (bf16)
// C[M][N] (f32) = A[M][K] (bf16 row-major) x Bt[N][K] (bf16 = B transposed).
// 256 threads (8 waves), 128x128 block tile, wave tile 32x64 (2x4 WMMA accums),
// async global->LDS tile copies, double-buffered, branch-free steady state.
__global__ __launch_bounds__(256) void gemm_bf16_nt(
    const unsigned short* __restrict__ A,
    const unsigned short* __restrict__ Bt,
    float* __restrict__ C, int M, int N, int K) {
  (void)M;
  __shared__ __align__(32) unsigned short lA[2][128 * 32];  // 2 x 8KB
  __shared__ __align__(32) unsigned short lB[2][128 * 32];  // 2 x 8KB
  const int tid = threadIdx.x, lane = tid & 31, wave = tid >> 5;
  const int wr = (wave >> 1) * 32;   // 0,32,64,96
  const int wc = (wave & 1) * 64;    // 0,64
  const size_t mBase = (size_t)blockIdx.y * 128;
  const size_t nBase = (size_t)blockIdx.x * 128;
  const int r = lane & 15, hi = lane >> 4;

  // per-thread tile-copy assignment: 512 uint4 per matrix tile / 256 threads
  const int idx0 = tid, idx1 = tid + 256;
  const int rA0 = idx0 >> 2, qA0 = (idx0 & 3) * 8;
  const int rA1 = idx1 >> 2, qA1 = (idx1 & 3) * 8;
  const unsigned short* gA0 = A + (mBase + rA0) * K + qA0;
  const unsigned short* gA1 = A + (mBase + rA1) * K + qA1;
  const unsigned short* gB0 = Bt + (nBase + rA0) * K + qA0;
  const unsigned short* gB1 = Bt + (nBase + rA1) * K + qA1;
  const unsigned laBase = lds_u32(&lA[0][0]);
  const unsigned lbBase = lds_u32(&lB[0][0]);
  const unsigned la0 = laBase + idx0 * 16, la1 = laBase + idx1 * 16;
  const unsigned lb0 = lbBase + idx0 * 16, lb1 = lbBase + idx1 * 16;

  // issue the next 32-K tile from the current pointers, then advance by 64B
  auto issue = [&](int b) {
    const unsigned bo = (unsigned)b * (128 * 32 * 2);
    async_cp16(la0 + bo, gA0);
    async_cp16(la1 + bo, gA1);
    async_cp16(lb0 + bo, gB0);
    async_cp16(lb1 + bo, gB1);
    gA0 += 32; gA1 += 32; gB0 += 32; gB1 += 32;
  };

  AccF acc[2][4];
#pragma unroll
  for (int a = 0; a < 2; ++a)
#pragma unroll
    for (int b = 0; b < 4; ++b) acc[a][b].v = (f32x8){0, 0, 0, 0, 0, 0, 0, 0};

  auto compute = [&](int b) {
    Frag16 afrag[2], bfrag[4];
#pragma unroll
    for (int mi = 0; mi < 2; ++mi) {  // A frag: K chunks hi*8 and 16+hi*8
      const unsigned short* p = &lA[b][(wr + mi * 16 + r) * 32];
      afrag[mi].q[0] = *(const uint4*)(p + hi * 8);
      afrag[mi].q[1] = *(const uint4*)(p + 16 + hi * 8);
    }
#pragma unroll
    for (int ni = 0; ni < 4; ++ni) {  // B frag: K chunks hi*8 and 16+hi*8
      const unsigned short* p = &lB[b][(wc + ni * 16 + r) * 32];
      bfrag[ni].q[0] = *(const uint4*)(p + hi * 8);
      bfrag[ni].q[1] = *(const uint4*)(p + 16 + hi * 8);
    }
#pragma unroll
    for (int mi = 0; mi < 2; ++mi)
#pragma unroll
      for (int ni = 0; ni < 4; ++ni)
        acc[mi][ni].v = WMMA_BF16(afrag[mi].bf, bfrag[ni].bf, acc[mi][ni].v);
  };

  issue(0);  // prologue: tile 0 -> buffer 0
  int buf = 0;
  const int iters = K / 32;
  for (int it = 0; it < iters - 1; ++it, buf ^= 1) {
    issue(buf ^ 1);                                  // prefetch next tile
    asm volatile("s_wait_asynccnt 4" ::: "memory");  // current tile landed
    __syncthreads();
    compute(buf);
    __syncthreads();  // all waves done reading buf before it is refilled
  }
  asm volatile("s_wait_asynccnt 0" ::: "memory");    // last tile landed
  __syncthreads();
  compute(buf);

  const int rbase = (lane < 16) ? 0 : 8;
#pragma unroll
  for (int mi = 0; mi < 2; ++mi)
#pragma unroll
    for (int ni = 0; ni < 4; ++ni)
#pragma unroll
      for (int i = 0; i < 8; ++i)
        C[(mBase + wr + mi * 16 + i + rbase) * N + nBase + wc + ni * 16 + r] =
            acc[mi][ni].e[i];
}

// -------------------------------------------------- RoPE + pack Q/K/V (bf16)
// qkv: [S][3*D] f32.  Qb,Kb: [S][D] bf16 (rotated).  Vt: [D][S] bf16 (V^T).
__global__ __launch_bounds__(64) void rope_pack(
    const float* __restrict__ qkv, unsigned short* __restrict__ Qb,
    unsigned short* __restrict__ Kb, unsigned short* __restrict__ Vt) {
  const int Dm = 4096, S = 2048;
  int h = blockIdx.x, s = blockIdx.y, j = threadIdx.x;  // j = pair index 0..63
  const float* base = qkv + (size_t)s * (3 * Dm) + h * 128 + 2 * j;
  float q0 = base[0], q1 = base[1];
  float k0 = base[Dm], k1 = base[Dm + 1];
  float v0 = base[2 * Dm], v1 = base[2 * Dm + 1];
  float freq = __powf(10000.0f, -(float)(2 * j) * (1.0f / 128.0f));
  float sn, cs;
  __sincosf((float)s * freq, &sn, &cs);
  size_t o = (size_t)s * Dm + h * 128 + 2 * j;
  Qb[o]     = f2bf_u(q0 * cs - q1 * sn);
  Qb[o + 1] = f2bf_u(q0 * sn + q1 * cs);
  Kb[o]     = f2bf_u(k0 * cs - k1 * sn);
  Kb[o + 1] = f2bf_u(k0 * sn + k1 * cs);
  Vt[(size_t)(h * 128 + 2 * j) * S + s]     = f2bf_u(v0);
  Vt[(size_t)(h * 128 + 2 * j + 1) * S + s] = f2bf_u(v1);
}

// ----------------------------------------------- attention, 1 wave per block
// grid (H, S/16). Two-pass streaming softmax; Ctx: [S][D] bf16 ("bqhd").
__global__ __launch_bounds__(32) void attn_fwd(
    const unsigned short* __restrict__ Qb, const unsigned short* __restrict__ Kb,
    const unsigned short* __restrict__ Vt, unsigned short* __restrict__ Ctx) {
  const int S = 2048, Dm = 4096;
  const float scale = 0.08838834764831845f;  // 1/sqrt(128)
  int h = blockIdx.x, qb = blockIdx.y, lane = threadIdx.x;
  __shared__ __align__(32) float lds_p[16 * 32];
  __shared__ float lds_m[16], lds_l[16];

  const int r = lane & 15, hi = lane >> 4;
  const int rbase = (lane < 16) ? 0 : 8;

  // resident Q fragments: 4 k-steps over HD=128
  Frag16 qf[4];
  {
    const unsigned short* qrow = Qb + (size_t)(qb * 16 + r) * Dm + h * 128;
#pragma unroll
    for (int kk = 0; kk < 4; ++kk) {
      qf[kk].q[0] = *(const uint4*)(qrow + kk * 32 + hi * 8);
      qf[kk].q[1] = *(const uint4*)(qrow + kk * 32 + 16 + hi * 8);
    }
  }

  // ---- pass 1: streaming row max m and denom l over all keys
  float m_run = -3.4e38f, l_run = 0.0f;
  for (int key0 = 0; key0 < S; key0 += 16) {
    AccF s;
    s.v = (f32x8){0, 0, 0, 0, 0, 0, 0, 0};
    const unsigned short* krow = Kb + (size_t)(key0 + r) * Dm + h * 128;
#pragma unroll
    for (int kk = 0; kk < 4; ++kk) {
      Frag16 kf;
      kf.q[0] = *(const uint4*)(krow + kk * 32 + hi * 8);
      kf.q[1] = *(const uint4*)(krow + kk * 32 + 16 + hi * 8);
      s.v = WMMA_BF16(qf[kk].bf, kf.bf, s.v);
    }
#pragma unroll
    for (int i = 0; i < 8; ++i) lds_p[(i + rbase) * 32 + r] = s.e[i] * scale;
    __syncthreads();
    if (lane < 16) {
      float tmax = -3.4e38f;
#pragma unroll
      for (int j = 0; j < 16; ++j) tmax = fmaxf(tmax, lds_p[lane * 32 + j]);
      float mnew = fmaxf(m_run, tmax);
      float sum = 0.0f;
#pragma unroll
      for (int j = 0; j < 16; ++j) sum += __expf(lds_p[lane * 32 + j] - mnew);
      l_run = l_run * __expf(m_run - mnew) + sum;
      m_run = mnew;
    }
    __syncthreads();
  }
  if (lane < 16) { lds_m[lane] = m_run; lds_l[lane] = l_run; }
  __syncthreads();
  const float m_r   = lds_m[r];
  const float inv_l = 1.0f / lds_l[r];

  // ---- pass 2: recompute scores, normalize, P.V accumulation
  AccF o[8];
#pragma unroll
  for (int ni = 0; ni < 8; ++ni) o[ni].v = (f32x8){0, 0, 0, 0, 0, 0, 0, 0};

  for (int key0 = 0; key0 < S; key0 += 32) {
#pragma unroll
    for (int hv = 0; hv < 2; ++hv) {
      AccF s;
      s.v = (f32x8){0, 0, 0, 0, 0, 0, 0, 0};
      const unsigned short* krow =
          Kb + (size_t)(key0 + hv * 16 + r) * Dm + h * 128;
#pragma unroll
      for (int kk = 0; kk < 4; ++kk) {
        Frag16 kf;
        kf.q[0] = *(const uint4*)(krow + kk * 32 + hi * 8);
        kf.q[1] = *(const uint4*)(krow + kk * 32 + 16 + hi * 8);
        s.v = WMMA_BF16(qf[kk].bf, kf.bf, s.v);
      }
#pragma unroll
      for (int i = 0; i < 8; ++i)
        lds_p[(i + rbase) * 32 + hv * 16 + r] = s.e[i] * scale;
    }
    __syncthreads();

    // P as bf16 A-fragment: row r, keys hi*8..+7 and 16+hi*8..+7
    Frag16 pf;
#pragma unroll
    for (int jj = 0; jj < 8; ++jj) {
      pf.us[jj] =
          f2bf_u(__expf(lds_p[r * 32 + hi * 8 + jj] - m_r) * inv_l);
      pf.us[8 + jj] =
          f2bf_u(__expf(lds_p[r * 32 + 16 + hi * 8 + jj] - m_r) * inv_l);
    }
#pragma unroll
    for (int ni = 0; ni < 8; ++ni) {  // V^T rows are B fragments
      Frag16 vf;
      const unsigned short* vrow =
          Vt + (size_t)(h * 128 + ni * 16 + r) * S + key0;
      vf.q[0] = *(const uint4*)(vrow + hi * 8);
      vf.q[1] = *(const uint4*)(vrow + 16 + hi * 8);
      o[ni].v = WMMA_BF16(pf.bf, vf.bf, o[ni].v);
    }
    __syncthreads();
  }

  // store context (bf16) in [S][D] layout, col = h*128 + d
#pragma unroll
  for (int ni = 0; ni < 8; ++ni)
#pragma unroll
    for (int i = 0; i < 8; ++i)
      Ctx[(size_t)(qb * 16 + i + rbase) * Dm + h * 128 + ni * 16 + r] =
          f2bf_u(o[ni].e[i]);
}

// ---------------------------------------------------------------------------

extern "C" void kernel_launch(void* const* d_in, const int* in_sizes, int n_in,
                              void* d_out, int out_size, void* d_ws,
                              size_t ws_size, hipStream_t stream) {
  (void)in_sizes; (void)n_in; (void)out_size; (void)ws_size;
  const float* hs   = (const float*)d_in[0];  // [1,2048,4096] f32
  const float* wqkv = (const float*)d_in[1];  // [4096,12288]  f32
  const float* wo   = (const float*)d_in[2];  // [4096,4096]   f32
  float* out = (float*)d_out;                 // [1,2048,4096] f32

  const size_t S = 2048, Dm = 4096, H = 32;
  char* ws = (char*)d_ws;
  size_t off = 0;
  auto take = [&](size_t bytes) -> void* {
    void* p = ws + off;
    off += (bytes + 255) & ~(size_t)255;
    return p;
  };
  unsigned short* Xb    = (unsigned short*)take(S * Dm * 2);        //  16 MB
  unsigned short* Wqkvt = (unsigned short*)take(3 * Dm * Dm * 2);   // 100 MB
  unsigned short* Wot   = (unsigned short*)take(Dm * Dm * 2);       //  33 MB
  float*          QKV   = (float*)take(S * 3 * Dm * 4);             // 100 MB
  unsigned short* Qb    = (unsigned short*)take(S * Dm * 2);
  unsigned short* Kb    = (unsigned short*)take(S * Dm * 2);
  unsigned short* Vt    = (unsigned short*)take(Dm * S * 2);
  unsigned short* Ctx   = (unsigned short*)take(S * Dm * 2);

  cvt_f32_bf16<<<2048, 256, 0, stream>>>(hs, Xb, (long)(S * Dm));
  transp_f32_bf16<<<dim3(3 * Dm / 32, Dm / 32), dim3(32, 8), 0, stream>>>(
      wqkv, Wqkvt, (int)Dm, (int)(3 * Dm));
  transp_f32_bf16<<<dim3(Dm / 32, Dm / 32), dim3(32, 8), 0, stream>>>(
      wo, Wot, (int)Dm, (int)Dm);
  gemm_bf16_nt<<<dim3(3 * Dm / 128, S / 128), 256, 0, stream>>>(
      Xb, Wqkvt, QKV, (int)S, (int)(3 * Dm), (int)Dm);
  rope_pack<<<dim3(H, S), 64, 0, stream>>>(QKV, Qb, Kb, Vt);
  attn_fwd<<<dim3(H, S / 16), 32, 0, stream>>>(Qb, Kb, Vt, Ctx);
  gemm_bf16_nt<<<dim3(Dm / 128, S / 128), 256, 0, stream>>>(
      Ctx, Wot, out, (int)S, (int)Dm, (int)Dm);
}
